// OmegaRule_91147795955943
// MI455X (gfx1250) — matrix operand order repaired and checked by reading, not compiled, not run on hardware
//
#include <hip/hip_runtime.h>

// Omega Rule (windowed) for MI455X gfx1250.
// bf16 WMMA (v_wmma_f32_16x16x32_bf16) with f32 accumulation, 3-phase plan:
//  P1: per-window Skk (bf16) / Svk (f32) rank-512 updates   [parallel, 1024 WGs]
//  P2: row-strip-parallel M recurrence; M strip in LDS (f32); Skk chunks staged
//      by the Tensor Data Mover (tensor_load_to_lds), double-buffered, exploiting
//      Skk's symmetry (K^T diag(g) K) so no transpose is needed [16 WGs x 128 steps]
//  P3: out = q @ M_snapshot^T; bf16 snapshots staged by TDM   [parallel, 1024 WGs]

typedef __bf16 bf16_t;
typedef bf16_t v16bf __attribute__((ext_vector_type(16)));
typedef float  v8f   __attribute__((ext_vector_type(8)));
typedef unsigned int v4u __attribute__((ext_vector_type(4)));
typedef int v4i __attribute__((ext_vector_type(4)));
typedef int v8i __attribute__((ext_vector_type(8)));

#define B_   8
#define L_   8192
#define D_   256
#define W_   64
#define NW_  128
#define KC_  32

// ---- WMMA operand gathers (wave32 layouts per CDNA5 ISA 7.12.2) -------------

// A matrix 16x32 bf16, row-major source A[16][lda], K-chunk at cols 0..31.
__device__ __forceinline__ v16bf load_A_bf(const bf16_t* A, int lda, int lane) {
  const int m = lane & 15, kh = lane >> 4;
  const bf16_t* row = A + m * lda;
  v16bf a;
#pragma unroll
  for (int j = 0; j < 8; ++j) {
    const int kb = ((j & 4) << 2) + 8 * kh + 2 * (j & 3);
    a[2 * j]     = row[kb];
    a[2 * j + 1] = row[kb + 1];
  }
  return a;
}

// Same, but source is f32 (convert to bf16 at gather time).
__device__ __forceinline__ v16bf load_A_f32(const float* A, int lda, int lane) {
  const int m = lane & 15, kh = lane >> 4;
  const float* row = A + m * lda;
  v16bf a;
#pragma unroll
  for (int j = 0; j < 8; ++j) {
    const int kb = ((j & 4) << 2) + 8 * kh + 2 * (j & 3);
    a[2 * j]     = (bf16_t)row[kb];
    a[2 * j + 1] = (bf16_t)row[kb + 1];
  }
  return a;
}

// B matrix 32x16 bf16 from Bt[N][ldb] (contiguous along K -> ds_load_b128).
__device__ __forceinline__ v16bf load_B_bf(const bf16_t* Bt, int ldb, int lane) {
  const int nn = lane & 15, kh = lane >> 4;
  const bf16_t* col = Bt + nn * ldb + 8 * kh;
  v16bf b;
#pragma unroll
  for (int j = 0; j < 16; ++j) b[j] = col[j + (j & 8)];
  return b;
}

__device__ __forceinline__ v8f wmma_bf16(v16bf a, v16bf b, v8f c) {
  return __builtin_amdgcn_wmma_f32_16x16x32_bf16(false, a, false, b,
                                                 (short)0, c, false, false);
}

// ---- TDM: async 2D tile copy, 256 rows x 32 bf16, row stride 256 elements ---
// D# per CDNA5 ISA 8.3/8.4. Issue from one wave only; TDM ignores EXEC.
__device__ __forceinline__ void tdm_load_tile_256x32(const bf16_t* src,
                                                     void* lds_dst) {
  const unsigned long long ga = (unsigned long long)(uintptr_t)src;
  const unsigned int la = (unsigned int)(uintptr_t)lds_dst;  // LDS = low 32 bits
  v4u g0;
  g0[0] = 1u;                                   // count=1, user descriptor
  g0[1] = la;                                   // lds_addr (bytes)
  g0[2] = (unsigned int)ga;                     // global_addr[31:0]
  g0[3] = (unsigned int)((ga >> 32) & 0x01FFFFFFull) | 0x80000000u; // [56:32]|type=2
  const unsigned int td = 1u << 30;             // huge tensor dims: no OOB clip
  v8i g1;
  g1[0] = 0x00010000;                           // data_size=1 (2-byte elements)
  g1[1] = (int)((td & 0xFFFFu) << 16);          // abar_addr=0 | tensor_dim0.lo16
  g1[2] = (int)(((td >> 16) & 0xFFFFu) | ((td & 0xFFFFu) << 16)); // d0.hi|d1.lo
  g1[3] = (int)(((td >> 16) & 0xFFFFu) | (32u << 16));  // d1.hi | tile_dim0=32
  g1[4] = 256;                                  // tile_dim1=256, tile_dim2=0
  g1[5] = 256;                                  // tensor_dim0_stride.lo32 = 256
  g1[6] = 0;                                    // stride0.hi16 | stride1.lo16
  g1[7] = 0;                                    // stride1.hi32
  v4i gz = {0, 0, 0, 0};
#if defined(__clang_major__) && (__clang_major__ >= 23)
  v8i gz8 = {0, 0, 0, 0, 0, 0, 0, 0};
  __builtin_amdgcn_tensor_load_to_lds(g0, g1, gz, gz, gz8, 0);
#else
  __builtin_amdgcn_tensor_load_to_lds(g0, g1, gz, gz, 0);
#endif
}

// ---- Phase 1: Skk[n] (bf16) and Svk[n] (f32) --------------------------------
// grid = NW*8; block owns a 32-row output strip d0..d0+31 of both matrices.
__global__ __launch_bounds__(256) void omega_phase1(
    const float* __restrict__ keys, const float* __restrict__ values,
    const float* __restrict__ gam, bf16_t* __restrict__ Skk,
    float* __restrict__ Svk) {
  __shared__ bf16_t kT[D_][KC_];    // kT[e][r]   : B operand (all 256 cols)
  __shared__ bf16_t gkT[KC_][KC_];  // gkT[d'][r] : A operand for Skk (strip)
  __shared__ bf16_t gvT[KC_][KC_];  // gvT[o'][r] : A operand for Svk (strip)

  const int n = blockIdx.x >> 3;
  const int s = blockIdx.x & 7;
  const int d0 = s * KC_;
  const int t = threadIdx.x;
  const int wave = t >> 5, lane = t & 31;

  v8f acc[8];
#pragma unroll
  for (int i = 0; i < 8; ++i)
#pragma unroll
    for (int j = 0; j < 8; ++j) acc[i][j] = 0.0f;

  const int row  = t >> 3;        // 0..31  chunk-local row
  const int col0 = (t & 7) * 32;  // 0..224 column group

  for (int c = 0; c < 16; ++c) {  // 16 chunks of 32 rows cover B*W = 512
    const int bb = c >> 1;
    const int w0 = (c & 1) * 32;
    const size_t lbase = (size_t)bb * L_ + (size_t)n * W_ + w0;

    const float* ksrc = keys + (lbase + row) * D_ + col0;
    float kv[32];
#pragma unroll
    for (int i = 0; i < 8; ++i) {
      const float4 f4 = ((const float4*)ksrc)[i];
      kv[4 * i + 0] = f4.x; kv[4 * i + 1] = f4.y;
      kv[4 * i + 2] = f4.z; kv[4 * i + 3] = f4.w;
    }
#pragma unroll
    for (int i = 0; i < 32; ++i) kT[col0 + i][row] = (bf16_t)kv[i];

    if ((t & 7) == s) {  // this thread's column group == strip: build g*k
      const float g = gam[lbase + row];
#pragma unroll
      for (int i = 0; i < 32; ++i) gkT[i][row] = (bf16_t)(g * kv[i]);
    }
    if (t < 32) {  // v strip, scaled by g, transposed
      const float g = gam[lbase + t];
      const float* vsrc = values + (lbase + t) * D_ + d0;
#pragma unroll
      for (int i = 0; i < 8; ++i) {
        const float4 f4 = ((const float4*)vsrc)[i];
        gvT[4 * i + 0][t] = (bf16_t)(g * f4.x);
        gvT[4 * i + 1][t] = (bf16_t)(g * f4.y);
        gvT[4 * i + 2][t] = (bf16_t)(g * f4.z);
        gvT[4 * i + 3][t] = (bf16_t)(g * f4.w);
      }
    }
    __syncthreads();

#pragma unroll
    for (int i = 0; i < 8; ++i) {
      const int tid = wave * 8 + i;
      const int trow = (tid >> 4) & 1;
      const bf16_t* Ap = (tid < 32) ? &gkT[trow * 16][0] : &gvT[trow * 16][0];
      const int tcol = tid & 15;
      const v16bf a  = load_A_bf(Ap, KC_, lane);
      const v16bf bm = load_B_bf(&kT[tcol * 16][0], KC_, lane);
      acc[i] = wmma_bf16(a, bm, acc[i]);
    }
    __syncthreads();
  }

  const int mlo = (lane >> 4) * 8;
  const int nn  = lane & 15;
  const size_t base = (size_t)n * (D_ * D_);
#pragma unroll
  for (int i = 0; i < 8; ++i) {
    const int tid  = wave * 8 + i;
    const int trow = (tid >> 4) & 1;
    const int tcol = tid & 15;
    if (tid < 32) {
#pragma unroll
      for (int j = 0; j < 8; ++j) {
        const int d = d0 + trow * 16 + j + mlo;
        Skk[base + (size_t)d * D_ + tcol * 16 + nn] = (bf16_t)acc[i][j];
      }
    } else {
#pragma unroll
      for (int j = 0; j < 8; ++j) {
        const int o = d0 + trow * 16 + j + mlo;
        Svk[base + (size_t)o * D_ + tcol * 16 + nn] = acc[i][j];
      }
    }
  }
}

// ---- Phase 2: sequential recurrence, 16-row M strips, TDM-staged Skk --------
// Skk is symmetric, so the B-operand layout Bt[e][dd] = Skk[e][kc*32+dd] is a
// plain strided tile copy -> TDM. Double-buffered, pipelined with the WMMAs.
__global__ __launch_bounds__(256) void omega_phase2(
    const bf16_t* __restrict__ Skk, const float* __restrict__ Svk,
    const float* __restrict__ alpha_p, bf16_t* __restrict__ Msnap_bf) {
  __shared__ float  Ms[16][D_];       // M strip (f32 master copy)
  __shared__ bf16_t Bt[2][D_][KC_];   // double-buffered TDM destination

  const int o0 = blockIdx.x * 16;
  const int t = threadIdx.x;
  const int wave = t >> 5, lane = t & 31;
  const float a_sig = 1.0f / (1.0f + __expf(-alpha_p[0]));

  for (int i = t; i < 16 * D_; i += 256) (&Ms[0][0])[i] = 0.0f;

  const int mlo = (lane >> 4) * 8;
  const int nn  = lane & 15;

  if (wave == 0) tdm_load_tile_256x32(Skk, &Bt[0][0][0]);  // chunk (n=0,kc=0)

  v8f acc[2];
  for (int c = 0; c < NW_ * 8; ++c) {   // flattened (window, k-chunk)
    const int n = c >> 3, kc = c & 7, cb = c & 1;
    const size_t wbase = (size_t)n * (D_ * D_);

    __syncthreads();  // all waves done reading buffer cb^1 (and prior Ms update)
    if (wave == 0 && c + 1 < NW_ * 8) {
      const int n2 = (c + 1) >> 3, kc2 = (c + 1) & 7;
      tdm_load_tile_256x32(Skk + (size_t)n2 * (D_ * D_) + kc2 * KC_,
                           &Bt[cb ^ 1][0][0]);
    }
    if (c + 1 < NW_ * 8) __builtin_amdgcn_s_wait_tensorcnt(1);  // current done
    else                 __builtin_amdgcn_s_wait_tensorcnt(0);
    __syncthreads();  // publish Bt[cb] to all waves

    if (kc == 0) {
#pragma unroll
      for (int i = 0; i < 2; ++i)
#pragma unroll
        for (int j = 0; j < 8; ++j) acc[i][j] = 0.0f;
    }

    const v16bf a = load_A_f32(&Ms[0][kc * KC_], D_, lane);
#pragma unroll
    for (int i = 0; i < 2; ++i) {
      const int tcol = wave * 2 + i;
      const v16bf bm = load_B_bf(&Bt[cb][tcol * 16][0], KC_, lane);
      acc[i] = wmma_bf16(a, bm, acc[i]);
    }

    if (kc == 7) {  // end of window: M = a*M - M@Skk + Svk ; snapshot (bf16)
      __syncthreads();  // all A-reads of Ms complete before overwrite
#pragma unroll
      for (int i = 0; i < 2; ++i) {
        const int tcol = wave * 2 + i;
#pragma unroll
        for (int j = 0; j < 8; ++j) {
          const int m   = j + mlo;
          const int col = tcol * 16 + nn;
          const float mnew = a_sig * Ms[m][col] - acc[i][j]
                           + Svk[wbase + (size_t)(o0 + m) * D_ + col];
          Ms[m][col] = mnew;
          Msnap_bf[wbase + (size_t)(o0 + m) * D_ + col] = (bf16_t)mnew;
        }
      }
    }
  }
}

// ---- Phase 3: out = q @ Msnap^T, TDM-staged bf16 snapshots ------------------
// grid = NW*8; block = one (window, batch) pair = 64 query rows.
__global__ __launch_bounds__(256) void omega_phase3(
    const float* __restrict__ queries, const bf16_t* __restrict__ Msnap_bf,
    float* __restrict__ out) {
  __shared__ bf16_t Bt[2][D_][KC_];  // Msnap rows are already Bt-layout (o, d)

  const int n = blockIdx.x >> 3;
  const int b = blockIdx.x & 7;
  const int t = threadIdx.x;
  const int wave = t >> 5, lane = t & 31;
  const int trow  = wave >> 1;  // 4 row-tiles of 16 q rows
  const int chalf = wave & 1;   // 8 col-tiles per wave
  const size_t wbase = (size_t)n * (D_ * D_);
  const size_t qrow0 = (size_t)b * L_ + (size_t)n * W_;

  v8f acc[8];
#pragma unroll
  for (int i = 0; i < 8; ++i)
#pragma unroll
    for (int j = 0; j < 8; ++j) acc[i][j] = 0.0f;

  if (wave == 0) tdm_load_tile_256x32(Msnap_bf + wbase, &Bt[0][0][0]);

  for (int kc = 0; kc < 8; ++kc) {
    const int cb = kc & 1;
    __syncthreads();  // previous compute done -> safe to refill other buffer
    if (wave == 0 && kc < 7)
      tdm_load_tile_256x32(Msnap_bf + wbase + (kc + 1) * KC_, &Bt[cb ^ 1][0][0]);
    if (kc < 7) __builtin_amdgcn_s_wait_tensorcnt(1);
    else        __builtin_amdgcn_s_wait_tensorcnt(0);
    __syncthreads();  // publish Bt[cb]

    const v16bf a = load_A_f32(
        queries + (qrow0 + (size_t)trow * 16) * D_ + kc * KC_, D_, lane);
#pragma unroll
    for (int i = 0; i < 8; ++i) {
      const int tcol = chalf * 8 + i;
      const v16bf bm = load_B_bf(&Bt[cb][tcol * 16][0], KC_, lane);
      acc[i] = wmma_bf16(a, bm, acc[i]);
    }
  }

  const int mlo = (lane >> 4) * 8;
  const int nn  = lane & 15;
#pragma unroll
  for (int i = 0; i < 8; ++i) {
    const int tcol = chalf * 8 + i;
#pragma unroll
    for (int j = 0; j < 8; ++j) {
      const int r = trow * 16 + j + mlo;
      out[(qrow0 + r) * D_ + tcol * 16 + nn] = acc[i][j];
    }
  }
}

// ---- launch -----------------------------------------------------------------
extern "C" void kernel_launch(void* const* d_in, const int* in_sizes, int n_in,
                              void* d_out, int out_size, void* d_ws, size_t ws_size,
                              hipStream_t stream) {
  (void)in_sizes; (void)n_in; (void)out_size; (void)ws_size;
  const float* keys    = (const float*)d_in[0];
  const float* values  = (const float*)d_in[1];
  const float* queries = (const float*)d_in[2];
  const float* gammas  = (const float*)d_in[3];
  const float* alpha   = (const float*)d_in[4];

  char* ws = (char*)d_ws;                                     // layout:
  bf16_t* Skk      = (bf16_t*)ws;                             // 16 MB bf16
  float*  Svk      = (float*)(ws + (size_t)16 * 1024 * 1024); // 32 MB f32
  bf16_t* Msnap_bf = (bf16_t*)(ws + (size_t)48 * 1024 * 1024);// 16 MB bf16
  float*  out      = (float*)d_out;

  omega_phase1<<<dim3(NW_ * 8), dim3(256), 0, stream>>>(keys, values, gammas, Skk, Svk);
  omega_phase2<<<dim3(16),      dim3(256), 0, stream>>>(Skk, Svk, alpha, Msnap_bf);
  omega_phase3<<<dim3(NW_ * 8), dim3(256), 0, stream>>>(queries, Msnap_bf, out);
}
